// GAT_74251394613580
// MI455X (gfx1250) — matrix operand order repaired and checked by reading, not compiled, not run on hardware
//
#include <hip/hip_runtime.h>
#include <stddef.h>
#include <stdint.h>

// ---------------------------------------------------------------------------
// Types for WMMA (gfx1250, wave32)
// ---------------------------------------------------------------------------
typedef __attribute__((ext_vector_type(8)))  __bf16 v8bf;
typedef __attribute__((ext_vector_type(16))) __bf16 v16bf;
typedef __attribute__((ext_vector_type(8)))  float  v8f;

// ---------------------------------------------------------------------------
// Helpers
// ---------------------------------------------------------------------------
__device__ __forceinline__ unsigned short f2bf(float f) {
  unsigned u = __float_as_uint(f);
  unsigned r = u + 0x7FFFu + ((u >> 16) & 1u);   // round-to-nearest-even
  return (unsigned short)(r >> 16);
}

// order-preserving float<->uint key (for atomicMax on floats incl. negatives)
__device__ __forceinline__ unsigned fkey(float f) {
  unsigned b = __float_as_uint(f);
  return (b & 0x80000000u) ? ~b : (b | 0x80000000u);
}
__device__ __forceinline__ float funkey(unsigned k) {
  unsigned b = (k & 0x80000000u) ? (k & 0x7FFFFFFFu) : ~k;
  return __uint_as_float(b);
}

__device__ __forceinline__ void edge_pair(const int* __restrict__ ei, int e, int E,
                                          int& s, int& d) {
  if (e < E) { s = ei[e]; d = ei[E + e]; }       // edge_index rows [0]=src [1]=dst
  else       { s = e - E; d = s; }               // appended self-loops
}

// ---------------------------------------------------------------------------
// f32 -> bf16 conversion
// ---------------------------------------------------------------------------
__global__ void cvt_f32_bf16(const float* __restrict__ in,
                             unsigned short* __restrict__ out, size_t n) {
  size_t i = (size_t)blockIdx.x * blockDim.x + threadIdx.x;
  if (i < n) out[i] = f2bf(in[i]);
}

// ---------------------------------------------------------------------------
// WMMA GEMM:  C[M, NCOLS] = A[M, K](bf16) * W[NCOLS, K](bf16)^T   (f32 accum)
//
// Register-blocked MRx NT macro-tile per wave (MR*16 rows x NT*16 cols).
// Per k-step: issue all MR A-fragments + NT B-fragments, then MR*NT WMMAs,
// so loads clause together and overlap with the WMMA burst.
//
// A layout (16-bit A 16x32): lane (r,half) loads K-chunks {half*8..+7} and
// {16+half*8..+7} of row r.  B layout: lane (r,half) holds the same K-chunks
// {half*8..+7} and {16+half*8..+7} of column r (== row r of W, row-major [NCOLS,K]).
// ---------------------------------------------------------------------------
template <int K, int NCOLS, int MR, int NT>
__global__ __launch_bounds__(256) void gat_gemm_wmma(
    const unsigned short* __restrict__ A,
    const unsigned short* __restrict__ W,
    float* __restrict__ C,
    int MT16,      // number of 16-row tiles (M is a multiple of 16)
    int MBlocks) { // ceil(MT16 / MR)
  constexpr int CG = NCOLS / (16 * NT);  // column groups
  const int wave = blockIdx.x * (blockDim.x >> 5) + (threadIdx.x >> 5);
  const int lane = threadIdx.x & 31;
  const int half = lane >> 4;
  const int r    = lane & 15;
  const int rb   = wave / CG;
  const int cb   = wave % CG;
  if (rb >= MBlocks) return;  // wave-uniform

  const __bf16* Ab = (const __bf16*)A;
  const __bf16* Wb = (const __bf16*)W;

  // A row pointers (tail row-tiles clamped; stores are guarded below)
  const __bf16* arow[MR];
#pragma unroll
  for (int m = 0; m < MR; ++m) {
    int rt = rb * MR + m;
    int rtc = rt < MT16 ? rt : MT16 - 1;
    arow[m] = Ab + (size_t)(rtc * 16 + r) * K + half * 8;
  }
  const __bf16* wbase = Wb + (size_t)(cb * NT * 16 + r) * K + half * 8;

  v8f acc[MR][NT] = {};

  for (int kk = 0; kk < K; kk += 32) {
    v16bf av[MR];
#pragma unroll
    for (int m = 0; m < MR; ++m) {
      v8bf alo = *(const v8bf*)(arow[m] + kk);
      v8bf ahi = *(const v8bf*)(arow[m] + kk + 16);
      av[m] = __builtin_shufflevector(alo, ahi,
                0, 1, 2, 3, 4, 5, 6, 7, 8, 9, 10, 11, 12, 13, 14, 15);
    }
    v16bf bv[NT];
#pragma unroll
    for (int t = 0; t < NT; ++t) {
      v8bf blo = *(const v8bf*)(wbase + (size_t)t * 16 * K + kk);
      v8bf bhi = *(const v8bf*)(wbase + (size_t)t * 16 * K + kk + 16);
      bv[t] = __builtin_shufflevector(blo, bhi,
                0, 1, 2, 3, 4, 5, 6, 7, 8, 9, 10, 11, 12, 13, 14, 15);
    }
#pragma unroll
    for (int m = 0; m < MR; ++m) {
#pragma unroll
      for (int t = 0; t < NT; ++t) {
        acc[m][t] = __builtin_amdgcn_wmma_f32_16x16x32_bf16(
            /*neg_a=*/false, av[m], /*neg_b=*/false, bv[t],
            /*c_mod=*/(short)0, acc[m][t], /*reuse_a=*/false, /*reuse_b=*/false);
      }
    }
  }

#pragma unroll
  for (int m = 0; m < MR; ++m) {
    int rt = rb * MR + m;
    if (rt < MT16) {  // wave-uniform guard
#pragma unroll
      for (int t = 0; t < NT; ++t) {
#pragma unroll
        for (int v = 0; v < 8; ++v) {
          // C/D layout: VGPR v, lane L -> (M = v + 8*(L/16), N = L%16)
          C[(size_t)(rt * 16 + v + half * 8) * NCOLS + (cb * NT + t) * 16 + r] =
              acc[m][t][v];
        }
      }
    }
  }
}

// ---------------------------------------------------------------------------
// Per-node attention logits: als[n,h] = sum_c h[n,h,c]*a_src[h,c]  (same ald)
// One wave per (node, head).
// ---------------------------------------------------------------------------
__global__ void node_attn_kernel(const float* __restrict__ h,
                                 const float* __restrict__ as_,
                                 const float* __restrict__ ad_,
                                 float* __restrict__ als,
                                 float* __restrict__ ald,
                                 int N, int H, int C) {
  int w = (int)((blockIdx.x * blockDim.x + threadIdx.x) >> 5);
  int lane = threadIdx.x & 31;
  if (w >= N * H) return;  // wave-uniform (waves are 32-aligned)
  int n = w / H, hh = w % H;
  const float* hp = h + ((size_t)n * H + hh) * C;
  float ss = 0.f, sd = 0.f;
  for (int c = lane; c < C; c += 32) {
    float v = hp[c];
    ss += v * as_[hh * C + c];
    sd += v * ad_[hh * C + c];
  }
  for (int off = 16; off; off >>= 1) {
    ss += __shfl_xor(ss, off, 32);
    sd += __shfl_xor(sd, off, 32);
  }
  if (lane == 0) { als[w] = ss; ald[w] = sd; }
}

// ---------------------------------------------------------------------------
// Edge pass 1: segment max (per dst, per head) via order-key atomicMax
// ---------------------------------------------------------------------------
template <int H>
__global__ void edge_max_kernel(const int* __restrict__ ei, int E, int N,
                                const float* __restrict__ als,
                                const float* __restrict__ ald,
                                unsigned* __restrict__ emax) {
  long long idx = (long long)blockIdx.x * blockDim.x + threadIdx.x;
  long long total = (long long)(E + N) * H;
  if (idx >= total) return;
  int hh = (int)(idx % H);
  int e  = (int)(idx / H);
  int s, d;
  edge_pair(ei, e, E, s, d);
  float x = als[(size_t)s * H + hh] + ald[(size_t)d * H + hh];
  x = x > 0.f ? x : 0.2f * x;  // leaky_relu 0.2
  atomicMax(&emax[(size_t)d * H + hh], fkey(x));
}

// ---------------------------------------------------------------------------
// Edge pass 2: ex = exp(e - emax[dst]); denom[dst] += ex  (store ex per edge)
// ---------------------------------------------------------------------------
template <int H>
__global__ void edge_expsum_kernel(const int* __restrict__ ei, int E, int N,
                                   const float* __restrict__ als,
                                   const float* __restrict__ ald,
                                   const unsigned* __restrict__ emax,
                                   float* __restrict__ ex,
                                   float* __restrict__ den) {
  long long idx = (long long)blockIdx.x * blockDim.x + threadIdx.x;
  long long total = (long long)(E + N) * H;
  if (idx >= total) return;
  int hh = (int)(idx % H);
  int e  = (int)(idx / H);
  int s, d;
  edge_pair(ei, e, E, s, d);
  float x = als[(size_t)s * H + hh] + ald[(size_t)d * H + hh];
  x = x > 0.f ? x : 0.2f * x;
  float m = funkey(emax[(size_t)d * H + hh]);
  float v = __expf(x - m);
  ex[(size_t)e * H + hh] = v;
  atomicAdd(&den[(size_t)d * H + hh], v);
}

// ---------------------------------------------------------------------------
// Edge pass 3: agg[dst,h,c] += h[src,h,c] * (ex / (den[dst,h]+1e-16))
// One thread per (edge, head, channel).
// ---------------------------------------------------------------------------
template <int H, int C>
__global__ void edge_agg_kernel(const int* __restrict__ ei, int E, int N,
                                const float* __restrict__ h,
                                const float* __restrict__ ex,
                                const float* __restrict__ den,
                                float* __restrict__ agg) {
  long long idx = (long long)blockIdx.x * blockDim.x + threadIdx.x;
  long long total = (long long)(E + N) * H * C;
  if (idx >= total) return;
  int c = (int)(idx % C);
  long long t = idx / C;
  int hh = (int)(t % H);
  int e  = (int)(t / H);
  int s, d;
  edge_pair(ei, e, E, s, d);
  float alpha = ex[(size_t)e * H + hh] / (den[(size_t)d * H + hh] + 1e-16f);
  float val = h[((size_t)s * H + hh) * C + c] * alpha;
  atomicAdd(&agg[((size_t)d * H + hh) * C + c], val);
}

// ---------------------------------------------------------------------------
// bias + ELU + convert to bf16 (input of next layer's GEMM)
// ---------------------------------------------------------------------------
__global__ void bias_elu_bf16_kernel(const float* __restrict__ agg,
                                     const float* __restrict__ b,
                                     unsigned short* __restrict__ out,
                                     size_t n, int F) {
  size_t i = (size_t)blockIdx.x * blockDim.x + threadIdx.x;
  if (i >= n) return;
  float v = agg[i] + b[(int)(i % (size_t)F)];
  v = v > 0.f ? v : (__expf(v) - 1.f);  // elu, alpha=1
  out[i] = f2bf(v);
}

// ---------------------------------------------------------------------------
// Final: out = log_softmax(agg + b3) over 32 channels. One wave per node.
// ---------------------------------------------------------------------------
__global__ void final_logsoftmax_kernel(const float* __restrict__ agg,
                                        const float* __restrict__ b,
                                        float* __restrict__ out, int N) {
  int w = (int)((blockIdx.x * blockDim.x + threadIdx.x) >> 5);
  int lane = threadIdx.x & 31;
  if (w >= N) return;
  float v = agg[(size_t)w * 32 + lane] + b[lane];
  float m = v;
  for (int off = 16; off; off >>= 1) m = fmaxf(m, __shfl_xor(m, off, 32));
  float e = __expf(v - m);
  float s = e;
  for (int off = 16; off; off >>= 1) s += __shfl_xor(s, off, 32);
  out[(size_t)w * 32 + lane] = v - m - __logf(s);
}

// ---------------------------------------------------------------------------
// Host orchestration
// ---------------------------------------------------------------------------
static inline unsigned gdim(long long n, int b) {
  return (unsigned)((n + b - 1) / b);
}

extern "C" void kernel_launch(void* const* d_in, const int* in_sizes, int n_in,
                              void* d_out, int out_size, void* d_ws, size_t ws_size,
                              hipStream_t stream) {
  const float* x   = (const float*)d_in[0];
  const int*   ei  = (const int*)d_in[1];
  const float* W1  = (const float*)d_in[2];
  const float* a1s = (const float*)d_in[3];
  const float* a1d = (const float*)d_in[4];
  const float* b1  = (const float*)d_in[5];
  const float* W2  = (const float*)d_in[6];
  const float* a2s = (const float*)d_in[7];
  const float* a2d = (const float*)d_in[8];
  const float* b2  = (const float*)d_in[9];
  const float* W3  = (const float*)d_in[10];
  const float* a3s = (const float*)d_in[11];
  const float* a3d = (const float*)d_in[12];
  const float* b3  = (const float*)d_in[13];

  const int N = in_sizes[0] / 512;   // 50000
  const int E = in_sizes[1] / 2;     // 800000
  const int ET = E + N;              // with self-loops
  const int H = 4, C = 64, F = H * C;  // 256
  const int MT16 = N / 16;           // 3125 16-row tiles (exact)
  const int MR = 4;
  const int MB = (MT16 + MR - 1) / MR;  // 782 row-blocks

  // ---- workspace layout (reused across layers) ----
  char* ws = (char*)d_ws;
  size_t off = 0;
  auto take = [&](size_t bytes) -> void* {
    void* p = ws + off;
    off = (off + bytes + 255) & ~(size_t)255;
    return p;
  };
  unsigned short* XB  = (unsigned short*)take((size_t)N * 512 * 2); // bf16 GEMM input
  float*          HB  = (float*)take((size_t)N * F * 4);            // GEMM output h
  float*          AGG = (float*)take((size_t)N * F * 4);            // aggregation
  unsigned short* WB1 = (unsigned short*)take((size_t)256 * 512 * 2);
  unsigned short* WB2 = (unsigned short*)take((size_t)256 * 256 * 2);
  unsigned short* WB3 = (unsigned short*)take((size_t)32 * 256 * 2);
  float*          ALS = (float*)take((size_t)N * H * 4);
  float*          ALD = (float*)take((size_t)N * H * 4);
  unsigned*       EMX = (unsigned*)take((size_t)N * H * 4);
  float*          DEN = (float*)take((size_t)N * H * 4);
  float*          EX  = (float*)take((size_t)ET * H * 4);

  // ---- weight + input conversion (deterministic every call) ----
  cvt_f32_bf16<<<gdim((long long)N * 512, 256), 256, 0, stream>>>(x, XB, (size_t)N * 512);
  cvt_f32_bf16<<<gdim(256 * 512, 256), 256, 0, stream>>>(W1, WB1, (size_t)256 * 512);
  cvt_f32_bf16<<<gdim(256 * 256, 256), 256, 0, stream>>>(W2, WB2, (size_t)256 * 256);
  cvt_f32_bf16<<<gdim(32 * 256, 256), 256, 0, stream>>>(W3, WB3, (size_t)32 * 256);

  // ---- layer 1 : in 512 -> 4 x 64 (concat) ----
  {
    long long waves = (long long)MB * 4;  // 4 column groups of 64
    gat_gemm_wmma<512, 256, 4, 4><<<gdim(waves * 32, 256), 256, 0, stream>>>(
        XB, WB1, HB, MT16, MB);
  }
  node_attn_kernel<<<gdim((long long)N * H * 32, 256), 256, 0, stream>>>(
      HB, a1s, a1d, ALS, ALD, N, H, C);
  hipMemsetAsync(EMX, 0, (size_t)N * H * 4, stream);
  hipMemsetAsync(DEN, 0, (size_t)N * H * 4, stream);
  hipMemsetAsync(AGG, 0, (size_t)N * F * 4, stream);
  edge_max_kernel<4><<<gdim((long long)ET * H, 256), 256, 0, stream>>>(
      ei, E, N, ALS, ALD, EMX);
  edge_expsum_kernel<4><<<gdim((long long)ET * H, 256), 256, 0, stream>>>(
      ei, E, N, ALS, ALD, EMX, EX, DEN);
  edge_agg_kernel<4, 64><<<gdim((long long)ET * H * C, 256), 256, 0, stream>>>(
      ei, E, N, HB, EX, DEN, AGG);
  bias_elu_bf16_kernel<<<gdim((long long)N * F, 256), 256, 0, stream>>>(
      AGG, b1, XB, (size_t)N * F, F);

  // ---- layer 2 : in 256 -> 4 x 64 (concat) ----
  {
    long long waves = (long long)MB * 4;
    gat_gemm_wmma<256, 256, 4, 4><<<gdim(waves * 32, 256), 256, 0, stream>>>(
        XB, WB2, HB, MT16, MB);
  }
  node_attn_kernel<<<gdim((long long)N * H * 32, 256), 256, 0, stream>>>(
      HB, a2s, a2d, ALS, ALD, N, H, C);
  hipMemsetAsync(EMX, 0, (size_t)N * H * 4, stream);
  hipMemsetAsync(DEN, 0, (size_t)N * H * 4, stream);
  hipMemsetAsync(AGG, 0, (size_t)N * F * 4, stream);
  edge_max_kernel<4><<<gdim((long long)ET * H, 256), 256, 0, stream>>>(
      ei, E, N, ALS, ALD, EMX);
  edge_expsum_kernel<4><<<gdim((long long)ET * H, 256), 256, 0, stream>>>(
      ei, E, N, ALS, ALD, EMX, EX, DEN);
  edge_agg_kernel<4, 64><<<gdim((long long)ET * H * C, 256), 256, 0, stream>>>(
      ei, E, N, HB, EX, DEN, AGG);
  bias_elu_bf16_kernel<<<gdim((long long)N * F, 256), 256, 0, stream>>>(
      AGG, b2, XB, (size_t)N * F, F);

  // ---- layer 3 : in 256 -> 1 x 32 (mean over 1 head == identity) ----
  {
    long long waves = (long long)MB;  // single 32-wide column group (NT=2)
    gat_gemm_wmma<256, 32, 4, 2><<<gdim(waves * 32, 256), 256, 0, stream>>>(
        XB, WB3, HB, MT16, MB);
  }
  node_attn_kernel<<<gdim((long long)N * 1 * 32, 256), 256, 0, stream>>>(
      HB, a3s, a3d, ALS, ALD, N, 1, 32);
  hipMemsetAsync(EMX, 0, (size_t)N * 1 * 4, stream);
  hipMemsetAsync(DEN, 0, (size_t)N * 1 * 4, stream);
  hipMemsetAsync(AGG, 0, (size_t)N * 32 * 4, stream);
  edge_max_kernel<1><<<gdim((long long)ET, 256), 256, 0, stream>>>(
      ei, E, N, ALS, ALD, EMX);
  edge_expsum_kernel<1><<<gdim((long long)ET, 256), 256, 0, stream>>>(
      ei, E, N, ALS, ALD, EMX, EX, DEN);
  edge_agg_kernel<1, 32><<<gdim((long long)ET * 32, 256), 256, 0, stream>>>(
      ei, E, N, HB, EX, DEN, AGG);
  final_logsoftmax_kernel<<<gdim((long long)N * 32, 256), 256, 0, stream>>>(
      AGG, b3, (float*)d_out, N);

  (void)n_in; (void)out_size; (void)ws_size;
}
